// SSSM_10977936408691
// MI455X (gfx1250) — matrix operand order repaired and checked by reading, not compiled, not run on hardware
//
#include <hip/hip_runtime.h>
#include <stdint.h>

// ---------------------------------------------------------------------------
// Spectral SSM forward for MI455X (gfx1250, wave32, WMMA).
//  - FFT conv replaced by Toeplitz-bank GEMMs (48MB of bf16 filters -> L2 resident)
//  - All big GEMMs: bf16 WMMA 16x16x32, f32 accumulate, 128x128 block tiles,
//    8 waves/block, each wave 32x64 (2x4 wmma tiles).
//  - Fast-path GEMMs: double-buffered LDS pipeline staged with
//    global_load_async_to_lds_b128 (ASYNCcnt); next tile's DMA overlaps the
//    current tile's WMMA; one barrier per K-step. Guarded path only for
//    ragged FFN dims / AR row shifts.
//  - Sequential y-recurrence: 2 blocks x 512 threads, state in LDS, m_y in L2.
//  - Workspace requirement: ~200 MB.
// ---------------------------------------------------------------------------

typedef __bf16 bf16;
typedef __attribute__((ext_vector_type(16))) __bf16 bf16x16;
typedef __attribute__((ext_vector_type(8)))  __bf16 bf16x8;
typedef __attribute__((ext_vector_type(8)))  float  f32x8;
typedef int i32x4_vs __attribute__((__vector_size__(16)));

#define D_     512
#define SLEN   1024
#define BSZ_   2
#define ROWS   (BSZ_ * SLEN)     // 2048
#define NEIG   24
#define KD     (NEIG * D_)       // 12288
#define NLAYER 4
#define HDIM   1365
#define HPAD   1368              // multiple of 8 -> 16B-aligned bf16 rows

#if __has_builtin(__builtin_amdgcn_global_load_async_to_lds_b128)
#define HAVE_ASYNC_LDS 1
#else
#define HAVE_ASYNC_LDS 0
#endif

#if HAVE_ASYNC_LDS
__device__ __forceinline__ void async_copy16(void* dst_lds, const void* src_gbl) {
    __builtin_amdgcn_global_load_async_to_lds_b128(
        (__attribute__((address_space(1))) i32x4_vs*)(src_gbl),
        (__attribute__((address_space(3))) i32x4_vs*)(dst_lds), 0, 0);
}
__device__ __forceinline__ void wait_async0() {
#if __has_builtin(__builtin_amdgcn_s_wait_asynccnt)
    __builtin_amdgcn_s_wait_asynccnt(0);
#else
    asm volatile("s_wait_asynccnt 0" ::: "memory");
#endif
}
#endif

// ------------------------------ conversions --------------------------------

__global__ __launch_bounds__(256) void k_cvt(const float* __restrict__ in,
                                             bf16* __restrict__ out, long long n) {
    long long i = (long long)blockIdx.x * 256 + threadIdx.x;
    if (i < n) out[i] = (bf16)in[i];
}

// out[c*ldout + r] = in[r*C + c]   (fp32 -> bf16 transpose)
__global__ __launch_bounds__(256) void k_cvt_T(const float* __restrict__ in,
                                               bf16* __restrict__ out,
                                               int R, int C, int ldout) {
    long long i = (long long)blockIdx.x * 256 + threadIdx.x;
    if (i >= (long long)R * C) return;
    int r = (int)(i / C), c = (int)(i % C);
    out[(long long)c * ldout + r] = (bf16)in[i];
}

// BuT[tap][o*512+j] = m_u[(o*512+j)*3 + tap]   (one layer)
__global__ __launch_bounds__(256) void k_build_buT(const float* __restrict__ mu,
                                                   bf16* __restrict__ out) {
    int i = blockIdx.x * 256 + threadIdx.x;            // 3*512*512
    if (i >= 3 * D_ * D_) return;
    int tap = i / (D_ * D_);
    int rem = i % (D_ * D_);
    out[i] = (bf16)mu[(long long)rem * 3 + tap];
}

// T[k][t][tau] = (tau<=t) ? eig_vals[k]^0.25 * eig_vecs[t-tau][k] : 0
__global__ __launch_bounds__(256) void k_build_T(const float* __restrict__ ev,
                                                 const float* __restrict__ evec,
                                                 bf16* __restrict__ T) {
    int t = blockIdx.x, k = blockIdx.y;
    float sc = powf(fmaxf(ev[k], 0.f), 0.25f);
    bf16* row = T + ((long long)k * SLEN + t) * SLEN;
    for (int tau = threadIdx.x; tau < SLEN; tau += 256) {
        float v = (tau <= t) ? sc * evec[(long long)(t - tau) * NEIG + k] : 0.f;
        row[tau] = (bf16)v;
    }
}

// ------------------------------ WMMA GEMM ----------------------------------
// C[m,n] (+)= sum_k A[m,k] * BT[n,k]    (A row-major, B stored transposed)
// GUARDED=false: M%128==0, N%128==0, K%32==0, shift==0 -> async double-buffer
// pipeline, no bounds checks. GUARDED=true: full edge/shift handling.

__device__ __forceinline__ void stage8(bf16* dst, const bf16* src, bool rok, int kvalid) {
    if (rok && kvalid >= 8) {
        *(bf16x8*)dst = *(const bf16x8*)src;
    } else {
#pragma unroll
        for (int j = 0; j < 8; ++j)
            dst[j] = (rok && j < kvalid) ? src[j] : (bf16)0.0f;
    }
}

// one K=32 tile of WMMA from staged LDS buffers
__device__ __forceinline__ void wmma_compute(const bf16 (&Asb)[128][40],
                                             const bf16 (&Bsb)[128][40],
                                             f32x8 (&acc)[2][4],
                                             int wm, int wn, int half, int l15) {
    // B fragments: lane<16 -> column n, K0..15 ; lane>=16 -> K16..31
    bf16x16 bfrag[4];
#pragma unroll
    for (int ni = 0; ni < 4; ++ni) {
        const int brow = wn * 64 + ni * 16 + l15;
        const int kb2  = half * 16;
        bf16x8 b0 = *(const bf16x8*)&Bsb[brow][kb2];
        bf16x8 b1 = *(const bf16x8*)&Bsb[brow][kb2 + 8];
        bfrag[ni] = __builtin_shufflevector(b0, b1, 0, 1, 2, 3, 4, 5, 6, 7,
                                            8, 9, 10, 11, 12, 13, 14, 15);
    }
    // A fragments: lane<16 -> row m, K0..15 ; lane>=16 -> K16..31
#pragma unroll
    for (int mi = 0; mi < 2; ++mi) {
        const int arow = wm * 32 + mi * 16 + l15;
        const int kb   = half * 16;
        bf16x8 alo = *(const bf16x8*)&Asb[arow][kb];
        bf16x8 ahi = *(const bf16x8*)&Asb[arow][kb + 8];
        bf16x16 af = __builtin_shufflevector(alo, ahi, 0, 1, 2, 3, 4, 5, 6, 7,
                                             8, 9, 10, 11, 12, 13, 14, 15);
#pragma unroll
        for (int ni = 0; ni < 4; ++ni) {
            acc[mi][ni] = __builtin_amdgcn_wmma_f32_16x16x32_bf16(
                false, af, false, bfrag[ni], (short)0, acc[mi][ni], false, false);
        }
    }
}

// stage one interior K-tile (async if available, else sync vector copies)
__device__ __forceinline__ void stage_tile_fast(bf16 (&Asb)[128][40], bf16 (&Bsb)[128][40],
                                                const bf16* A, const bf16* BT,
                                                int lda, int ldbt, int m0, int n0,
                                                int kt, int rA, int kc) {
#pragma unroll
    for (int it = 0; it < 2; ++it) {
        int r = rA + it * 64;
        const bf16* sa = A  + (long long)(m0 + r) * lda  + kt + kc;
        const bf16* sb = BT + (long long)(n0 + r) * ldbt + kt + kc;
#if HAVE_ASYNC_LDS
        async_copy16(&Asb[r][kc], sa);   // global -> LDS, ASYNCcnt
        async_copy16(&Bsb[r][kc], sb);
#else
        *(bf16x8*)&Asb[r][kc] = *(const bf16x8*)sa;
        *(bf16x8*)&Bsb[r][kc] = *(const bf16x8*)sb;
#endif
    }
}

template <bool GUARDED, bool ACCUM, bool BF16OUT>
__global__ __launch_bounds__(256) void k_gemm(
    const bf16* __restrict__ A, const bf16* __restrict__ BT, void* __restrict__ Cv,
    int M, int N, int K, int lda, int ldbt, int ldc,
    long long sAzhi, long long sBzlo, long long sCzlo, long long sCzhi, int zdiv,
    int shift, int shiftMod,
    const float* __restrict__ bias, int biasMod, int ldbias) {
    __shared__ bf16 As[2][128][40];
    __shared__ bf16 Bs[2][128][40];

    const int z = blockIdx.z;
    const int zhi = z / zdiv, zlo = z % zdiv;
    A  += (long long)zhi * sAzhi;
    BT += (long long)zlo * sBzlo;
    const long long coff = (long long)zlo * sCzlo + (long long)zhi * sCzhi;

    const int m0 = blockIdx.y * 128, n0 = blockIdx.x * 128;
    const int tid  = threadIdx.x;
    const int lane = tid & 31, wave = tid >> 5;
    const int wm = wave >> 1, wn = wave & 1;          // 4x2 wave grid
    const int half = lane >> 4, l15 = lane & 15;

    f32x8 acc[2][4] = {};

    const int rA = tid >> 2;
    const int kc = (tid & 3) * 8;

    if (!GUARDED) {
        // -------- software-pipelined double buffer --------
        const int nt = K >> 5;
        stage_tile_fast(As[0], Bs[0], A, BT, lda, ldbt, m0, n0, 0, rA, kc);
        for (int i = 0; i < nt; ++i) {
#if HAVE_ASYNC_LDS
            wait_async0();          // my tile-i copies have landed
#endif
            __syncthreads();        // everyone's landed; prev buffer free
            const int cur = i & 1;
            if (i + 1 < nt)         // DMA tile i+1 while computing tile i
                stage_tile_fast(As[cur ^ 1], Bs[cur ^ 1], A, BT, lda, ldbt,
                                m0, n0, (i + 1) * 32, rA, kc);
            if (i + 2 < nt) {       // warm L2 two tiles ahead
                __builtin_prefetch(A  + (long long)(m0 + rA) * lda  + (i + 2) * 32 + kc, 0, 1);
                __builtin_prefetch(BT + (long long)(n0 + rA) * ldbt + (i + 2) * 32 + kc, 0, 1);
            }
            wmma_compute(As[cur], Bs[cur], acc, wm, wn, half, l15);
        }
    } else {
        // -------- conservative guarded path (ragged dims / AR shift) --------
        for (int kt = 0; kt < K; kt += 32) {
            const int krem = K - kt;
#pragma unroll
            for (int it = 0; it < 2; ++it) {
                int r  = rA + it * 64;
                int gm = m0 + r;
                bool rok = (gm < M) && ((gm % shiftMod) >= shift);
                stage8(&As[0][r][kc], A + (long long)(gm - shift) * lda + kt + kc, rok, krem - kc);
                int gn = n0 + r;
                stage8(&Bs[0][r][kc], BT + (long long)gn * ldbt + kt + kc, gn < N, krem - kc);
            }
            if (kt + 32 < K) {
                int gm = m0 + rA;
                if (gm < M && (gm % shiftMod) >= shift)
                    __builtin_prefetch(A + (long long)(gm - shift) * lda + kt + 32 + kc, 0, 1);
                int gn = n0 + rA;
                if (gn < N)
                    __builtin_prefetch(BT + (long long)gn * ldbt + kt + 32 + kc, 0, 1);
            }
            __syncthreads();
            wmma_compute(As[0], Bs[0], acc, wm, wn, half, l15);
            __syncthreads();
        }
    }

    float* Cf = (float*)Cv;
    bf16*  Cb = (bf16*)Cv;
#pragma unroll
    for (int mi = 0; mi < 2; ++mi)
#pragma unroll
        for (int ni = 0; ni < 4; ++ni)
#pragma unroll
            for (int g = 0; g < 8; ++g) {
                int mrow = m0 + wm * 32 + mi * 16 + half * 8 + g;
                int ncol = n0 + wn * 64 + ni * 16 + l15;
                if (!GUARDED || (mrow < M && ncol < N)) {
                    long long idx = coff + (long long)mrow * ldc + ncol;
                    float v = acc[mi][ni][g];
                    if (bias) v += bias[(long long)(mrow % biasMod) * ldbias + ncol];
                    if (ACCUM) v += Cf[idx];
                    if (BF16OUT) Cb[idx] = (bf16)v; else Cf[idx] = v;
                }
            }
}

// ------------------------------ pointwise ----------------------------------

// one block per row r: z = rms(x)*rn1 ; h = rms(x)*rn2 (bf16 row-major + bf16^T)
__global__ __launch_bounds__(256) void k_rmsnorm_dual(
    const float* __restrict__ x, const float* __restrict__ w1,
    const float* __restrict__ w2, float* __restrict__ z,
    bf16* __restrict__ h, bf16* __restrict__ hT) {
    __shared__ float red[256];
    int r = blockIdx.x, tid = threadIdx.x;
    int b = r >> 10, t = r & 1023;
    const float* xr = x + (long long)r * D_;
    float s = 0.f;
    for (int j = tid; j < D_; j += 256) { float v = xr[j]; s += v * v; }
    red[tid] = s; __syncthreads();
    for (int o = 128; o > 0; o >>= 1) { if (tid < o) red[tid] += red[tid + o]; __syncthreads(); }
    float scale = rsqrtf(red[0] * (1.0f / D_) + 1e-6f);
    for (int j = tid; j < D_; j += 256) {
        float v = xr[j] * scale;
        z[(long long)r * D_ + j] = v * w1[j];
        float hv = v * w2[j];
        h[(long long)r * D_ + j] = (bf16)hv;
        hT[((long long)b * D_ + j) * SLEN + t] = (bf16)hv;
    }
}

// y_t = M0 @ y_{t-1} + M1 @ y_{t-2} + delta_t  (2 blocks, 512 threads)
__global__ __launch_bounds__(512) void k_recur(
    const float* __restrict__ m_y, const float* __restrict__ delta,
    float* __restrict__ y, bf16* __restrict__ ybf) {
    __shared__ __align__(16) float y1[D_];
    __shared__ __align__(16) float y2[D_];
    int b = blockIdx.x, o = threadIdx.x;
    y1[o] = 0.f; y2[o] = 0.f; __syncthreads();
    const float4* M0 = (const float4*)(m_y + (long long)o * 1024);
    const float4* M1 = (const float4*)(m_y + (long long)o * 1024 + 512);
    for (int t = 0; t < SLEN; ++t) {
        const float4* Y1 = (const float4*)y1;
        const float4* Y2 = (const float4*)y2;
        float s = 0.f;
        for (int d4 = 0; d4 < D_ / 4; ++d4) {
            float4 a = M0[d4], v = Y1[d4];
            s += a.x * v.x + a.y * v.y + a.z * v.z + a.w * v.w;
            float4 c = M1[d4], w = Y2[d4];
            s += c.x * w.x + c.y * w.y + c.z * w.z + c.w * w.w;
        }
        long long idx = ((long long)b * SLEN + t) * D_ + o;
        float val = s + delta[idx];
        __syncthreads();
        y2[o] = y1[o];
        y1[o] = val;
        y[idx]   = val;
        ybf[idx] = (bf16)val;
        __syncthreads();
    }
}

__global__ __launch_bounds__(256) void k_swiglu(const float* __restrict__ g,
                                                const float* __restrict__ u,
                                                bf16* __restrict__ s, long long n) {
    long long i = (long long)blockIdx.x * 256 + threadIdx.x;
    if (i >= n) return;
    int c = (int)(i % HPAD);
    float out = 0.f;
    if (c < HDIM) {
        float a = g[i];
        out = (a / (1.f + __expf(-a))) * u[i];
    }
    s[i] = (bf16)out;
}

__global__ __launch_bounds__(256) void k_residual(const float* __restrict__ yv,
                                                  const float* __restrict__ ff,
                                                  const float* __restrict__ z,
                                                  float* __restrict__ x,
                                                  bf16* __restrict__ xbf, long long n) {
    long long i = (long long)blockIdx.x * 256 + threadIdx.x;
    if (i >= n) return;
    float v = yv[i] + ff[i] + z[i];
    x[i] = v;
    xbf[i] = (bf16)v;
}

__global__ __launch_bounds__(256) void k_loss_partial(const float* __restrict__ p,
                                                      const float* __restrict__ t,
                                                      float* __restrict__ part, long long n) {
    __shared__ float red[256];
    float s = 0.f;
    for (long long i = (long long)blockIdx.x * 256 + threadIdx.x; i < n;
         i += (long long)gridDim.x * 256) {
        float d = p[i] - t[i];
        s += d * d;
    }
    red[threadIdx.x] = s; __syncthreads();
    for (int o = 128; o > 0; o >>= 1) {
        if (threadIdx.x < o) red[threadIdx.x] += red[threadIdx.x + o];
        __syncthreads();
    }
    if (threadIdx.x == 0) part[blockIdx.x] = red[0];
}

__global__ __launch_bounds__(256) void k_loss_final(const float* __restrict__ part,
                                                    float* __restrict__ out, int nb, float invn) {
    __shared__ float red[256];
    float s = 0.f;
    for (int i = threadIdx.x; i < nb; i += 256) s += part[i];
    red[threadIdx.x] = s; __syncthreads();
    for (int o = 128; o > 0; o >>= 1) {
        if (threadIdx.x < o) red[threadIdx.x] += red[threadIdx.x + o];
        __syncthreads();
    }
    if (threadIdx.x == 0) out[0] = red[0] * invn;
}

// ------------------------------ launcher -----------------------------------

extern "C" void kernel_launch(void* const* d_in, const int* in_sizes, int n_in,
                              void* d_out, int out_size, void* d_ws, size_t ws_size,
                              hipStream_t stream) {
    (void)in_sizes; (void)n_in;
    const float* inputs  = (const float*)d_in[0];
    const float* targets = (const float*)d_in[1];
    const float* w_emb   = (const float*)d_in[2];
    const float* wpe     = (const float*)d_in[3];
    const float* eigval  = (const float*)d_in[4];
    const float* eigvec  = (const float*)d_in[5];
    const float* rn1     = (const float*)d_in[6];
    const float* rn2     = (const float*)d_in[7];
    const float* m_u     = (const float*)d_in[8];
    const float* m_phi   = (const float*)d_in[9];
    const float* m_y     = (const float*)d_in[10];
    const float* ffw1    = (const float*)d_in[11];
    const float* ffw2    = (const float*)d_in[12];
    const float* ffw3    = (const float*)d_in[13];
    const float* w_head  = (const float*)d_in[14];

    // workspace layout (~200 MB)
    char* base = (char*)d_ws;
    size_t off = 0;
    auto alloc = [&](size_t bytes) -> void* {
        off = (off + 255) & ~(size_t)255;
        void* p = base + off;
        off += bytes;
        return p;
    };
    bf16*  Tmat   = (bf16*)alloc((size_t)NEIG * SLEN * SLEN * 2);
    bf16*  wembT  = (bf16*)alloc((size_t)D_ * D_ * 2);
    bf16*  mphiT  = (bf16*)alloc((size_t)NLAYER * D_ * KD * 2);
    bf16*  buT    = (bf16*)alloc((size_t)NLAYER * 3 * D_ * D_ * 2);
    bf16*  w1T    = (bf16*)alloc((size_t)NLAYER * HDIM * D_ * 2);
    bf16*  w3T    = (bf16*)alloc((size_t)NLAYER * HDIM * D_ * 2);
    bf16*  w2T    = (bf16*)alloc((size_t)NLAYER * D_ * HPAD * 2);
    bf16*  wheadT = (bf16*)alloc((size_t)D_ * D_ * 2);
    bf16*  ainbf  = (bf16*)alloc((size_t)ROWS * D_ * 2);
    float* x      = (float*)alloc((size_t)ROWS * D_ * 4);
    bf16*  xbf    = (bf16*)alloc((size_t)ROWS * D_ * 2);
    float* zbuf   = (float*)alloc((size_t)ROWS * D_ * 4);
    bf16*  hbf    = (bf16*)alloc((size_t)ROWS * D_ * 2);
    bf16*  hbfT   = (bf16*)alloc((size_t)BSZ_ * D_ * SLEN * 2);
    bf16*  xt     = (bf16*)alloc((size_t)BSZ_ * SLEN * KD * 2);
    float* delta  = (float*)alloc((size_t)ROWS * D_ * 4);
    float* yv     = (float*)alloc((size_t)ROWS * D_ * 4);
    bf16*  ybf    = (bf16*)alloc((size_t)ROWS * D_ * 2);
    float* ffbuf  = (float*)alloc((size_t)ROWS * D_ * 4);
    float* lpart  = (float*)alloc(1024 * 4);
    if (off > ws_size) return;  // workspace too small: bail (defensive)

    // g/u/s alias the x_tilde region (xt is dead once delta GEMM has run,
    // and next layer's conv GEMM fully rewrites it)
    float* g   = (float*)xt;
    float* u   = (float*)((char*)xt + (size_t)ROWS * HPAD * 4);
    bf16*  sbf = (bf16*)((char*)xt + (size_t)ROWS * HPAD * 8);

    const long long Z0 = 0;
    auto blks = [](long long n) { return (unsigned)((n + 255) / 256); };

    // ---- weight conversion / Toeplitz bank build ----
    k_cvt_T<<<blks((long long)D_ * D_), 256, 0, stream>>>(w_emb, wembT, D_, D_, D_);
    k_cvt_T<<<blks((long long)D_ * D_), 256, 0, stream>>>(w_head, wheadT, D_, D_, D_);
    for (int i = 0; i < NLAYER; ++i) {
        k_cvt_T<<<blks((long long)KD * D_), 256, 0, stream>>>(
            m_phi + (size_t)i * KD * D_, mphiT + (size_t)i * D_ * KD, KD, D_, KD);
        k_cvt_T<<<blks((long long)D_ * HDIM), 256, 0, stream>>>(
            ffw1 + (size_t)i * D_ * HDIM, w1T + (size_t)i * HDIM * D_, D_, HDIM, D_);
        k_cvt_T<<<blks((long long)D_ * HDIM), 256, 0, stream>>>(
            ffw3 + (size_t)i * D_ * HDIM, w3T + (size_t)i * HDIM * D_, D_, HDIM, D_);
        k_cvt_T<<<blks((long long)HDIM * D_), 256, 0, stream>>>(
            ffw2 + (size_t)i * HDIM * D_, w2T + (size_t)i * D_ * HPAD, HDIM, D_, HPAD);
        k_build_buT<<<blks(3LL * D_ * D_), 256, 0, stream>>>(
            m_u + (size_t)i * D_ * D_ * 3, buT + (size_t)i * 3 * D_ * D_);
    }
    k_build_T<<<dim3(SLEN, NEIG), 256, 0, stream>>>(eigval, eigvec, Tmat);
    k_cvt<<<blks((long long)ROWS * D_), 256, 0, stream>>>(inputs, ainbf, (long long)ROWS * D_);

    // ---- embedding: x = inputs @ w_emb + wpe ----
    k_gemm<false, false, false><<<dim3(4, 16, 1), 256, 0, stream>>>(
        ainbf, wembT, x, ROWS, D_, D_, D_, D_, D_,
        Z0, Z0, Z0, Z0, 1, 0, 1, wpe, SLEN, D_);

    for (int i = 0; i < NLAYER; ++i) {
        // rmsnorms
        k_rmsnorm_dual<<<ROWS, 256, 0, stream>>>(x, rn1 + i * D_, rn2 + i * D_,
                                                 zbuf, hbf, hbfT);
        // spectral conv: x_tilde[b,t,k*512+j] = sum_tau T_k[t,tau] h[b,tau,j]
        k_gemm<false, false, true><<<dim3(4, 8, NEIG * BSZ_), 256, 0, stream>>>(
            Tmat, hbfT, xt, SLEN, D_, SLEN, SLEN, SLEN, KD,
            (long long)SLEN * SLEN,          // A advance per k
            (long long)D_ * SLEN,            // B advance per b
            (long long)SLEN * KD,            // C advance per b
            (long long)D_,                   // C advance per k
            BSZ_, 0, 1, nullptr, 1, 0);
        // delta = x_tilde @ m_phi
        k_gemm<false, false, false><<<dim3(4, 16, 1), 256, 0, stream>>>(
            xt, mphiT + (size_t)i * D_ * KD, delta, ROWS, D_, KD, KD, KD, D_,
            Z0, Z0, Z0, Z0, 1, 0, 1, nullptr, 1, 0);
        // AR taps: delta += h[t-tap] @ Bu_tap  (sequential launches -> no C race)
        for (int tap = 0; tap < 3; ++tap) {
            k_gemm<true, true, false><<<dim3(4, 16, 1), 256, 0, stream>>>(
                hbf, buT + ((size_t)i * 3 + tap) * D_ * D_, delta,
                ROWS, D_, D_, D_, D_, D_,
                Z0, Z0, Z0, Z0, 1, tap, SLEN, nullptr, 1, 0);
        }
        // sequential recurrence
        k_recur<<<BSZ_, D_, 0, stream>>>(m_y + (size_t)i * D_ * 2 * D_, delta, yv, ybf);
        // FFN (SwiGLU)
        k_gemm<true, false, false><<<dim3(11, 16, 1), 256, 0, stream>>>(
            ybf, w1T + (size_t)i * HDIM * D_, g, ROWS, HDIM, D_, D_, D_, HPAD,
            Z0, Z0, Z0, Z0, 1, 0, 1, nullptr, 1, 0);
        k_gemm<true, false, false><<<dim3(11, 16, 1), 256, 0, stream>>>(
            ybf, w3T + (size_t)i * HDIM * D_, u, ROWS, HDIM, D_, D_, D_, HPAD,
            Z0, Z0, Z0, Z0, 1, 0, 1, nullptr, 1, 0);
        k_swiglu<<<blks((long long)ROWS * HPAD), 256, 0, stream>>>(
            g, u, sbf, (long long)ROWS * HPAD);
        k_gemm<true, false, false><<<dim3(4, 16, 1), 256, 0, stream>>>(
            sbf, w2T + (size_t)i * D_ * HPAD, ffbuf, ROWS, D_, HDIM, HPAD, HPAD, D_,
            Z0, Z0, Z0, Z0, 1, 0, 1, nullptr, 1, 0);
        // x = y + ff + z
        k_residual<<<blks((long long)ROWS * D_), 256, 0, stream>>>(
            yv, ffbuf, zbuf, x, xbf, (long long)ROWS * D_);
    }

    // head + loss
    float* preds = (float*)d_out;
    k_gemm<false, false, false><<<dim3(4, 16, 1), 256, 0, stream>>>(
        xbf, wheadT, preds, ROWS, D_, D_, D_, D_, D_,
        Z0, Z0, Z0, Z0, 1, 0, 1, nullptr, 1, 0);
    const long long nelem = (long long)ROWS * D_;
    k_loss_partial<<<1024, 256, 0, stream>>>(preds, targets, lpart, nelem);
    k_loss_final<<<1, 256, 0, stream>>>(lpart, preds + (out_size - 1), 1024,
                                        1.0f / (float)nelem);
}
